// BasicBlock_38903813767380
// MI455X (gfx1250) — matrix operand order repaired and checked by reading, not compiled, not run on hardware
//
#include <hip/hip_runtime.h>

typedef __attribute__((ext_vector_type(16))) _Float16 v16h;
typedef __attribute__((ext_vector_type(8)))  float    v8f;

#define B_   32
#define C_   64
#define H_   56
#define W_   56
#define HW_  (H_*W_)        // 3136
#define NPIX (B_*HW_)       // 100352
#define KCH  18             // K chunks: 9 taps * 2 channel-halves (K=576, 32 per WMMA)
#define FRAG_HALFS (4*KCH*32*16)   // 36864 f16 per conv weight
#define ROWS 4              // output rows per block
#define LROWS (ROWS+2)      // staged input rows (halo)

// ---------------------------------------------------------------------------
// Convert OIHW fp32 weights into WMMA A-fragment order (f16):
// frag[((m_tile*18 + k_chunk)*32 + lane)*16 + j]
// A 16x32 order used here (same as B): lanes 0-15 hold K=0..15, lanes 16-31 K=16..31
// Global K ordering: k = tap*64 + channel  (tap = kh*3+kw)
// ---------------------------------------------------------------------------
__global__ void prep_weights(const float* __restrict__ w1,
                             const float* __restrict__ w2,
                             _Float16* __restrict__ f1,
                             _Float16* __restrict__ f2) {
  int idx = blockIdx.x * blockDim.x + threadIdx.x;
  if (idx >= 2 * FRAG_HALFS) return;
  const float* w = (idx < FRAG_HALFS) ? w1 : w2;
  _Float16*    f = (idx < FRAG_HALFS) ? f1 : f2;
  int r      = idx % FRAG_HALFS;
  int j      = r & 15;
  int lane   = (r >> 4) & 31;
  int kc     = (r >> 9) % KCH;
  int m_tile = r / (KCH * 512);
  int k;
  if (lane < 16) k = j;        // K 0..15
  else           k = j + 16;   // K 16..31
  int t  = kc >> 1;              // tap 0..8
  int hc = kc & 1;               // channel half
  int ch = hc * 32 + k;          // input channel 0..63
  int m  = m_tile * 16 + (lane & 15);
  int kh = t / 3, kw = t % 3;
  f[r] = (_Float16)w[((m * C_ + ch) * 3 + kh) * 3 + kw];
}

// ---------------------------------------------------------------------------
// Implicit-GEMM 3x3 conv via v_wmma_f32_16x16x32_f16.
// One workgroup per (batch, 4-output-row group). LDS stages 6 input rows
// (66 cols with zero pad) x 64 channels, channel-contiguous so each B
// fragment is 2 aligned ds_load_b128.
// 8 waves = 2 channel-halves (PM=2 M-tiles each) x 4 rows; each wave owns
// 8 accumulators (32M x 64N) -> each A frag feeds 4 WMMAs, each B frag 2.
// Optionally applies BN(+ReLU) of the *previous* layer while filling LDS.
// ---------------------------------------------------------------------------
__global__ __launch_bounds__(256) void conv3x3_bn_wmma(
    const float* __restrict__ src,       // NCHW fp32 activations
    const _Float16* __restrict__ wfrag,  // prepacked A fragments
    const float* __restrict__ bias,
    const float* __restrict__ bn_mean,
    const float* __restrict__ bn_rstd,
    const float* __restrict__ bn_g,
    const float* __restrict__ bn_b,
    int apply_bn,
    float* __restrict__ dst)             // NCHW fp32 conv output (+bias)
{
  __shared__ __align__(32) _Float16 lds[LROWS * 66 * 64];

  int b  = blockIdx.x / (H_ / ROWS);
  int h0 = (blockIdx.x % (H_ / ROWS)) * ROWS;

  // Cooperative fill: input rows h0-1 .. h0+4, cols -1..64 (zero padded), 64 ch.
  for (int rr = 0; rr < LROWS; ++rr) {
    int hin = h0 + rr - 1;
    bool hok = (hin >= 0) && (hin < H_);
    for (int i = threadIdx.x; i < 66 * 64; i += 256) {
      int c   = i & 63;
      int col = i >> 6;
      int win = col - 1;
      float v = 0.0f;
      if (hok && win >= 0 && win < W_) {
        v = src[((size_t)(b * C_ + c) * H_ + hin) * W_ + win];
        if (apply_bn) {
          v = (v - bn_mean[c]) * bn_rstd[c] * bn_g[c] + bn_b[c];
          v = v > 0.0f ? v : 0.0f;
        }
      }
      lds[rr * (66 * 64) + i] = (_Float16)v;
    }
  }
  __syncthreads();

  int lane = threadIdx.x & 31;
  int wv   = threadIdx.x >> 5;
  int mgrp = wv & 1;            // channel half: channels [mgrp*32, mgrp*32+32)
  int row  = wv >> 1;           // output row within block: 0..3
  int lmod = lane & 15;
  int chalf = (lane < 16) ? 0 : 16;

  v8f acc[8] = {};              // [mt*4 + ns]: mt in {0,1}, ns in {0..3}
  const _Float16* wb0 = wfrag + ((size_t)((mgrp * 2 + 0) * KCH) * 32 + lane) * 16;
  const _Float16* wb1 = wfrag + ((size_t)((mgrp * 2 + 1) * KCH) * 32 + lane) * 16;

  #pragma unroll
  for (int t = 0; t < 9; ++t) {
    int kh = t / 3, kw = t % 3;
    int ldsrow = row + kh;                       // input row h0+row+kh-1
    #pragma unroll
    for (int hc = 0; hc < 2; ++hc) {
      int kc = t * 2 + hc;
      v16h a0 = *(const v16h*)(wb0 + (size_t)kc * 32 * 16);
      v16h a1 = *(const v16h*)(wb1 + (size_t)kc * 32 * 16);
      int cb = hc * 32 + chalf;                  // channel base (16-aligned)
      #pragma unroll
      for (int ns = 0; ns < 4; ++ns) {
        int col = ns * 16 + lmod + kw;           // 0..65
        v16h bf = *(const v16h*)(&lds[((size_t)ldsrow * 66 + col) * 64 + cb]);
        acc[ns]     = __builtin_amdgcn_wmma_f32_16x16x32_f16(
                          false, a0, false, bf, (short)0, acc[ns],     false, false);
        acc[4 + ns] = __builtin_amdgcn_wmma_f32_16x16x32_f16(
                          false, a1, false, bf, (short)0, acc[4 + ns], false, false);
      }
    }
  }

  // C/D layout: VGPR r -> M = r (lanes 0-15) or r+8 (lanes 16-31), N = lane%16
  int h = h0 + row;
  int moff = (lane < 16) ? 0 : 8;
  #pragma unroll
  for (int mt = 0; mt < 2; ++mt) {
    #pragma unroll
    for (int ns = 0; ns < 4; ++ns) {
      int pix = ns * 16 + lmod;
      if (pix < W_) {
        #pragma unroll
        for (int r = 0; r < 8; ++r) {
          int ch = (mgrp * 2 + mt) * 16 + moff + r;
          dst[((size_t)(b * C_ + ch) * H_ + h) * (size_t)W_ + pix] =
              acc[mt * 4 + ns][r] + bias[ch];
        }
      }
    }
  }
}

// ---------------------------------------------------------------------------
// Per-channel batch statistics (training-mode BN, biased variance).
// One block per channel; data is L2-resident after the conv.
// ---------------------------------------------------------------------------
__global__ __launch_bounds__(256) void bn_stats(const float* __restrict__ y,
                                                float* __restrict__ mean,
                                                float* __restrict__ rstd) {
  __shared__ float s_sum[256];
  __shared__ float s_sq[256];
  int ch = blockIdx.x;
  float s = 0.0f, s2 = 0.0f;
  for (int p = threadIdx.x; p < NPIX; p += 256) {
    int bb = p / HW_;
    int hw = p % HW_;
    float v = y[((size_t)bb * C_ + ch) * HW_ + hw];
    s += v; s2 += v * v;
  }
  s_sum[threadIdx.x] = s;
  s_sq[threadIdx.x]  = s2;
  __syncthreads();
  for (int off = 128; off > 0; off >>= 1) {
    if (threadIdx.x < off) {
      s_sum[threadIdx.x] += s_sum[threadIdx.x + off];
      s_sq[threadIdx.x]  += s_sq[threadIdx.x + off];
    }
    __syncthreads();
  }
  if (threadIdx.x == 0) {
    float m   = s_sum[0] / (float)NPIX;
    float var = s_sq[0] / (float)NPIX - m * m;
    mean[ch] = m;
    rstd[ch] = rsqrtf(var + 1e-5f);
  }
}

// ---------------------------------------------------------------------------
// out = relu(bn2(y2) + x), in-place on d_out (each thread owns one element).
// ---------------------------------------------------------------------------
__global__ void finalize_k(float* __restrict__ y2,
                           const float* __restrict__ x,
                           const float* __restrict__ mean,
                           const float* __restrict__ rstd,
                           const float* __restrict__ g,
                           const float* __restrict__ be,
                           int n) {
  int idx = blockIdx.x * blockDim.x + threadIdx.x;
  if (idx >= n) return;
  int c = (idx / HW_) & 63;
  float v = y2[idx];
  v = (v - mean[c]) * rstd[c] * g[c] + be[c] + x[idx];
  y2[idx] = v > 0.0f ? v : 0.0f;
}

// ---------------------------------------------------------------------------
extern "C" void kernel_launch(void* const* d_in, const int* in_sizes, int n_in,
                              void* d_out, int out_size, void* d_ws, size_t ws_size,
                              hipStream_t stream) {
  const float* x   = (const float*)d_in[0];
  const float* w1  = (const float*)d_in[1];
  const float* b1  = (const float*)d_in[2];
  const float* g1  = (const float*)d_in[3];
  const float* be1 = (const float*)d_in[4];
  const float* w2  = (const float*)d_in[5];
  const float* b2  = (const float*)d_in[6];
  const float* g2  = (const float*)d_in[7];
  const float* be2 = (const float*)d_in[8];
  float* out = (float*)d_out;

  char* ws = (char*)d_ws;
  _Float16* f1 = (_Float16*)ws;                       // 73,728 B
  _Float16* f2 = (_Float16*)(ws + 73728);             // 73,728 B
  float* mean1 = (float*)(ws + 147456);               // 4 x 64 floats of stats
  float* rstd1 = mean1 + 64;
  float* mean2 = rstd1 + 64;
  float* rstd2 = mean2 + 64;
  float* y1    = (float*)(ws + 148480);               // 25,690,112 B

  prep_weights<<<(2 * FRAG_HALFS + 255) / 256, 256, 0, stream>>>(w1, w2, f1, f2);

  int nblk = B_ * (H_ / ROWS);                        // 448

  // conv1 (+bias) -> y1  (apply_bn=0: BN pointers unused)
  conv3x3_bn_wmma<<<nblk, 256, 0, stream>>>(
      x, f1, b1, mean1, rstd1, g1, be1, 0, y1);

  bn_stats<<<C_, 256, 0, stream>>>(y1, mean1, rstd1);

  // conv2 (+bias) over relu(bn1(y1)) -> d_out (holds pre-BN2 conv output)
  conv3x3_bn_wmma<<<nblk, 256, 0, stream>>>(
      y1, f2, b2, mean1, rstd1, g1, be1, 1, out);

  bn_stats<<<C_, 256, 0, stream>>>(out, mean2, rstd2);

  int n = B_ * C_ * HW_;
  finalize_k<<<(n + 255) / 256, 256, 0, stream>>>(out, x, mean2, rstd2, g2, be2, n);
}
